// GroupedQueryAttention_8358006358227
// MI455X (gfx1250) — compile-verified, matrix operand order checked by reading, not run on hardware
//
#include <hip/hip_runtime.h>
#include <hip/hip_bf16.h>

typedef __bf16 bf16_t;
typedef bf16_t v16bf __attribute__((ext_vector_type(16)));
typedef bf16_t v8bf  __attribute__((ext_vector_type(8)));
typedef float  v8f   __attribute__((ext_vector_type(8)));
typedef unsigned int v4u __attribute__((ext_vector_type(4)));
typedef int v8i __attribute__((ext_vector_type(8)));
typedef int v4i __attribute__((ext_vector_type(4)));

#define B_    4
#define S_    1024
#define DM_   1024
#define KV_   256
#define DH_   32
#define MROWS (B_ * S_)   // 4096 flattened rows

// ---------------------------------------------------------------------------
// Fragment helpers for V_WMMA_F32_16X16X32_BF16 (wave32).
// 16-bit A-matrix 16x32 layout (ISA 7.12.2): lanes 0-15 & 16-31 both hold
// rows M=0..15; per lane, VGPR0-3 hold K = 8*half + {0..7}, VGPR4-7 hold
// K = 16 + 8*half + {0..7}.  Each group is 8 contiguous bf16 = one 16B load.
// B-matrix mirrors this with N in place of M, so row-major [N,K] loads the
// same way.  Works for both global and LDS base pointers.
// ---------------------------------------------------------------------------
static __device__ __forceinline__ v16bf load_frag(const bf16_t* base, int ld,
                                                  int row, int kbase, int lane) {
  const int half = (lane >> 4) & 1;
  const bf16_t* p = base + (size_t)row * ld + kbase + 8 * half;
  v8bf lo = *(const v8bf*)(p);
  v8bf hi = *(const v8bf*)(p + 16);
  v16bf r;
#pragma unroll
  for (int i = 0; i < 8; ++i) { r[i] = lo[i]; r[i + 8] = hi[i]; }
  return r;
}

static __device__ __forceinline__ v8f wmma_bf16(v16bf a, v16bf b, v8f c) {
  // (neg_a, A, neg_b, B, c_mod, C, reuse_a, reuse_b)
  return __builtin_amdgcn_wmma_f32_16x16x32_bf16(false, a, false, b,
                                                 (short)0, c, false, false);
}

// ---------------------------------------------------------------------------
// Tensor Data Mover: 2D tile (height x width bf16, row stride ld elems) from
// global memory into LDS at byte offset lds_off.  Descriptor per ISA
// CDNA5 ISA:
//   group0: [1:0]=count=1, [63:32]=lds_addr, [120:64]=global_addr, type=2
//   group1: data_size=1(2B), tensor_dim0/1 big (tiles always in-bounds),
//           tile_dim0=width, tile_dim1=height, tensor_dim0_stride=ld
// This toolchain (clang-23 / therock-10.0) exposes the 6-arg builtin:
//   (v4u g0, v8i g1, v4i g2, v4i g3, v8i g4, i32 cpol)
// Tracked with TENSORcnt; completes in order with other TDM ops of the wave.
// ---------------------------------------------------------------------------
static __device__ __forceinline__ void tdm_load_2d(unsigned int lds_off,
                                                   const bf16_t* gptr,
                                                   unsigned int width,
                                                   unsigned int height,
                                                   unsigned int ld) {
  unsigned long long ga = (unsigned long long)(uintptr_t)gptr;
  const unsigned int td0 = 1u << 20;   // tensor dims >> tile dims: no clipping
  const unsigned int td1 = 1u << 20;
  v4u g0;
  g0[0] = 1u;                                       // count=1 (valid user D#)
  g0[1] = lds_off;                                  // lds_addr
  g0[2] = (unsigned int)(ga & 0xffffffffu);         // global_addr[31:0]
  g0[3] = (unsigned int)((ga >> 32) & 0x01ffffffu)  // global_addr[56:32]
        | (2u << 30);                               // type=2 ("image")
  v8i g1;
  g1[0] = (int)(1u << 16);                          // data_size=1 -> 2 bytes
  g1[1] = (int)((td0 & 0xffffu) << 16);             // tensor_dim0[15:0]
  g1[2] = (int)((td0 >> 16) | ((td1 & 0xffffu) << 16));
  g1[3] = (int)((td1 >> 16) | (width << 16));       // tile_dim0
  g1[4] = (int)(height & 0xffffu);                  // tile_dim1
  g1[5] = (int)ld;                                  // tensor_dim0_stride lo32
  g1[6] = 0;                                        // stride hi + dim1_stride
  g1[7] = 0;
  v4i g2 = {0, 0, 0, 0};                            // 2D tensor: groups 2/3
  v4i g3 = {0, 0, 0, 0};                            // unused (zeros)
  v8i g4 = {0, 0, 0, 0, 0, 0, 0, 0};
  __builtin_amdgcn_tensor_load_to_lds(g0, g1, g2, g3, g4, 0);
}

// ---------------------------------------------------------------------------
// f32 -> bf16 conversion
// ---------------------------------------------------------------------------
__global__ void cvt_f32_bf16(const float* __restrict__ src,
                             bf16_t* __restrict__ dst, int n) {
  int i = blockIdx.x * blockDim.x + threadIdx.x;
  if (i < n) dst[i] = (bf16_t)src[i];
}

// ---------------------------------------------------------------------------
// WMMA GEMM with TDM double-buffered LDS staging.
// C[M,N] = A[M,K] @ B[N,K]^T (+bias).  One block = 64x64 C tile, 4 waves,
// each wave a 32x32 quadrant (4 accumulators, 4 WMMA per 32-deep K step).
// Wave 0 drives the TDM pipeline: issue next slab -> s_wait_tensorcnt ->
// barrier -> all waves compute from LDS (ds_load_b128 fragments).
// ---------------------------------------------------------------------------
__global__ __launch_bounds__(128) void gemm_bf16_tdm(const bf16_t* __restrict__ A,
                                                     const bf16_t* __restrict__ Bm,
                                                     bf16_t* __restrict__ Cb,
                                                     float* __restrict__ Cf,
                                                     const float* __restrict__ bias,
                                                     int M, int N, int K) {
  __shared__ alignas(16) bf16_t smA[2][64 * 32];
  __shared__ alignas(16) bf16_t smB[2][64 * 32];
  const int lane = threadIdx.x & 31;
  const int wave = threadIdx.x >> 5;
  const int tilesN = N >> 6;
  const int mB = (blockIdx.x / tilesN) * 64;
  const int nB = (blockIdx.x % tilesN) * 64;
  const int mo = (wave >> 1) * 32;          // quadrant offset inside 64x64
  const int no = (wave & 1) * 32;
  const int col = lane & 15;

  v8f acc00 = {}, acc01 = {}, acc10 = {}, acc11 = {};
  const int KT = K >> 5;

  if (wave == 0) {
    tdm_load_2d((unsigned int)(uintptr_t)&smA[0][0], A + (size_t)mB * K, 32, 64, K);
    tdm_load_2d((unsigned int)(uintptr_t)&smB[0][0], Bm + (size_t)nB * K, 32, 64, K);
  }
  int buf = 0;
  for (int kt = 0; kt < KT; ++kt) {
    if (wave == 0) {
      if (kt + 1 < KT) {
        const int k = (kt + 1) << 5;
        tdm_load_2d((unsigned int)(uintptr_t)&smA[buf ^ 1][0],
                    A + (size_t)mB * K + k, 32, 64, K);
        tdm_load_2d((unsigned int)(uintptr_t)&smB[buf ^ 1][0],
                    Bm + (size_t)nB * K + k, 32, 64, K);
        __builtin_amdgcn_s_wait_tensorcnt(2);  // in-order: current slab done
      } else {
        __builtin_amdgcn_s_wait_tensorcnt(0);
      }
    }
    __syncthreads();
    const bf16_t* At = smA[buf];
    const bf16_t* Bt = smB[buf];
    v16bf a0 = load_frag(At, 32, mo + col,      0, lane);
    v16bf a1 = load_frag(At, 32, mo + 16 + col, 0, lane);
    v16bf b0 = load_frag(Bt, 32, no + col,      0, lane);
    v16bf b1 = load_frag(Bt, 32, no + 16 + col, 0, lane);
    acc00 = wmma_bf16(a0, b0, acc00);
    acc01 = wmma_bf16(a0, b1, acc01);
    acc10 = wmma_bf16(a1, b0, acc10);
    acc11 = wmma_bf16(a1, b1, acc11);
    __syncthreads();                           // slab consumed before reuse
    buf ^= 1;
  }

  const int half = (lane >> 4) & 1;
#pragma unroll
  for (int v = 0; v < 8; ++v) {
    const int r = v + 8 * half;
    const int m0 = mB + mo + r, m1 = mB + mo + 16 + r;
    const int n0 = nB + no + col, n1 = nB + no + 16 + col;
    float v00 = acc00[v], v01 = acc01[v], v10 = acc10[v], v11 = acc11[v];
    if (bias) {
      v00 += bias[n0]; v01 += bias[n1];
      v10 += bias[n0]; v11 += bias[n1];
    }
    if (Cf) {
      Cf[(size_t)m0 * N + n0] = v00; Cf[(size_t)m0 * N + n1] = v01;
      Cf[(size_t)m1 * N + n0] = v10; Cf[(size_t)m1 * N + n1] = v11;
    }
    if (Cb) {
      Cb[(size_t)m0 * N + n0] = (bf16_t)v00; Cb[(size_t)m0 * N + n1] = (bf16_t)v01;
      Cb[(size_t)m1 * N + n0] = (bf16_t)v10; Cb[(size_t)m1 * N + n1] = (bf16_t)v11;
    }
  }
}

// ---------------------------------------------------------------------------
// Interleaved RoPE in-place on bf16 rows; theta denominator fixed at
// D_MODEL/2 = 512 for both q (pairs=512) and k (pairs=128), per reference.
// ---------------------------------------------------------------------------
__global__ void rope_bf16(bf16_t* __restrict__ t, int rows, int rowlen, int pairs) {
  int idx = blockIdx.x * blockDim.x + threadIdx.x;
  if (idx >= rows * pairs) return;
  int row = idx / pairs;
  int i   = idx - row * pairs;
  int pos = row & (S_ - 1);
  float theta = __powf(10000.0f, -(float)i * (1.0f / 512.0f));
  float ang = (float)pos * theta;
  float c, s;
  __sincosf(ang, &s, &c);
  bf16_t* p = t + (size_t)row * rowlen + 2 * i;
  float xe = (float)p[0], xo = (float)p[1];
  p[0] = (bf16_t)(xe * c - xo * s);
  p[1] = (bf16_t)(xo * c + xe * s);
}

// ---------------------------------------------------------------------------
// v [b][s][c] -> vt [b][c][s]   (c = kv_head*32 + feat), so PV's B-operand
// is contiguous along keys per lane.
// ---------------------------------------------------------------------------
__global__ void vtrans(const bf16_t* __restrict__ vb, bf16_t* __restrict__ vt) {
  int idx = blockIdx.x * blockDim.x + threadIdx.x;
  const int total = B_ * S_ * KV_;
  if (idx >= total) return;
  int b   = idx / (S_ * KV_);
  int rem = idx - b * (S_ * KV_);
  int s   = rem / KV_;
  int c   = rem - s * KV_;
  vt[((size_t)b * KV_ + c) * S_ + s] = vb[idx];
}

// ---------------------------------------------------------------------------
// Flash attention: one wave per (batch, qhead, 16-query tile).
// Per 32-key block: 2 WMMA for S = Q K^T, online softmax with shfl_xor row
// reductions over the 16-lane halves, P re-fragmented via LDS (wave-local,
// DS ops are in-order within a wave), 2 WMMA for O += P V.
// ---------------------------------------------------------------------------
__global__ __launch_bounds__(128) void attn_fwd(const bf16_t* __restrict__ qb,
                                                const bf16_t* __restrict__ kb,
                                                const bf16_t* __restrict__ vt,
                                                bf16_t* __restrict__ attn) {
  __shared__ alignas(16) bf16_t pshare[4][16 * 32];
  const int lane = threadIdx.x & 31;
  const int wave = threadIdx.x >> 5;
  const int wid  = blockIdx.x * 4 + wave;
  const int qt  = wid & 63;                  // 64 query tiles of 16
  const int h   = (wid >> 6) & 31;           // 32 query heads
  const int b   = wid >> 11;                 // 4 batches
  const int kvh = h >> 2;                    // GQA: 4 q-heads share a kv-head
  const int q0  = qt * 16;
  const int col  = lane & 15;
  const int half = (lane >> 4) & 1;

  const bf16_t* Qb = qb + (size_t)b * S_ * DM_ + (size_t)h * DH_;     // ld=1024
  const bf16_t* Kb = kb + (size_t)b * S_ * KV_ + (size_t)kvh * DH_;   // ld=256
  const bf16_t* Vt = vt + ((size_t)b * KV_ + (size_t)kvh * DH_) * S_; // ld=1024
  bf16_t* P = pshare[wave];

  v16bf qf = load_frag(Qb, DM_, q0 + col, 0, lane);

  v8f o0 = {}, o1 = {};
  float rmax[8], rsum[8];
#pragma unroll
  for (int v = 0; v < 8; ++v) { rmax[v] = -1e30f; rsum[v] = 0.0f; }
  const float scale = 0.1767766952966369f;   // 1/sqrt(32)

  for (int k0 = 0; k0 <= q0 + 15; k0 += 32) {
    if (k0 + 32 <= q0 + 15) {                // prefetch next key/value block
      __builtin_prefetch(Kb + (size_t)(k0 + 32 + col) * KV_, 0, 3);
      __builtin_prefetch(Vt + (size_t)col * S_ + k0 + 32, 0, 3);
    }
    v16bf kf0 = load_frag(Kb, KV_, k0 + col,      0, lane);
    v16bf kf1 = load_frag(Kb, KV_, k0 + 16 + col, 0, lane);
    v8f s0 = {}, s1 = {};
    s0 = wmma_bf16(qf, kf0, s0);
    s1 = wmma_bf16(qf, kf1, s1);

#pragma unroll
    for (int v = 0; v < 8; ++v) {
      const int qrow = q0 + v + 8 * half;
      float a0 = s0[v] * scale;              // key = k0 + col
      float a1 = s1[v] * scale;              // key = k0 + 16 + col
      if (k0 + col      > qrow) a0 = -1e30f; // causal mask
      if (k0 + 16 + col > qrow) a1 = -1e30f;
      float mx = fmaxf(a0, a1);
#pragma unroll
      for (int m = 1; m < 16; m <<= 1) mx = fmaxf(mx, __shfl_xor(mx, m, 32));
      const float nmax = fmaxf(rmax[v], mx);
      const float corr = __expf(rmax[v] - nmax);
      const float p0 = __expf(a0 - nmax);
      const float p1 = __expf(a1 - nmax);
      float ps = p0 + p1;
#pragma unroll
      for (int m = 1; m < 16; m <<= 1) ps += __shfl_xor(ps, m, 32);
      rsum[v] = rsum[v] * corr + ps;
      rmax[v] = nmax;
      o0[v] *= corr;
      o1[v] *= corr;
      P[(v + 8 * half) * 32 + col]      = (bf16_t)p0;   // D-layout -> LDS
      P[(v + 8 * half) * 32 + 16 + col] = (bf16_t)p1;
    }

    // Reload P in A-matrix layout (same wave: DS ops complete in order)
    v16bf pf  = load_frag(P, 32, col, 0, lane);
    v16bf vf0 = load_frag(Vt, S_, col,      k0, lane);  // feat 0..15
    v16bf vf1 = load_frag(Vt, S_, 16 + col, k0, lane);  // feat 16..31
    o0 = wmma_bf16(pf, vf0, o0);
    o1 = wmma_bf16(pf, vf1, o1);
  }

#pragma unroll
  for (int v = 0; v < 8; ++v) {
    const int qrow = q0 + v + 8 * half;
    const float inv = 1.0f / rsum[v];
    const size_t base = ((size_t)b * S_ + qrow) * DM_ + (size_t)h * DH_;
    attn[base + col]      = (bf16_t)(o0[v] * inv);
    attn[base + 16 + col] = (bf16_t)(o1[v] * inv);
  }
}

// ---------------------------------------------------------------------------
extern "C" void kernel_launch(void* const* d_in, const int* in_sizes, int n_in,
                              void* d_out, int out_size, void* d_ws, size_t ws_size,
                              hipStream_t stream) {
  const float* x  = (const float*)d_in[0];
  const float* Wq = (const float*)d_in[1];
  const float* Wk = (const float*)d_in[2];
  const float* Wv = (const float*)d_in[3];
  const float* Wo = (const float*)d_in[4];
  const float* bo = (const float*)d_in[5];
  // d_in[6] = mask: causal, recomputed in-kernel.

  char* ws = (char*)d_ws;
  size_t off = 0;
  auto alloc = [&](size_t bytes) -> char* {
    char* p = ws + off;
    off += (bytes + 255) & ~(size_t)255;
    return p;
  };
  bf16_t* xb  = (bf16_t*)alloc((size_t)MROWS * DM_ * 2);
  bf16_t* Wqb = (bf16_t*)alloc((size_t)DM_ * DM_ * 2);
  bf16_t* Wkb = (bf16_t*)alloc((size_t)KV_ * DM_ * 2);
  bf16_t* Wvb = (bf16_t*)alloc((size_t)KV_ * DM_ * 2);
  bf16_t* Wob = (bf16_t*)alloc((size_t)DM_ * DM_ * 2);
  bf16_t* qbw = (bf16_t*)alloc((size_t)MROWS * DM_ * 2);
  bf16_t* kbw = (bf16_t*)alloc((size_t)MROWS * KV_ * 2);
  bf16_t* vbw = (bf16_t*)alloc((size_t)MROWS * KV_ * 2);
  bf16_t* vtw = (bf16_t*)alloc((size_t)MROWS * KV_ * 2);
  bf16_t* atw = (bf16_t*)alloc((size_t)MROWS * DM_ * 2);

  // 1) convert inputs to bf16
  {
    int n = MROWS * DM_;
    cvt_f32_bf16<<<(n + 255) / 256, 256, 0, stream>>>(x, xb, n);
    n = DM_ * DM_;
    cvt_f32_bf16<<<(n + 255) / 256, 256, 0, stream>>>(Wq, Wqb, n);
    cvt_f32_bf16<<<(n + 255) / 256, 256, 0, stream>>>(Wo, Wob, n);
    n = KV_ * DM_;
    cvt_f32_bf16<<<(n + 255) / 256, 256, 0, stream>>>(Wk, Wkb, n);
    cvt_f32_bf16<<<(n + 255) / 256, 256, 0, stream>>>(Wv, Wvb, n);
  }

  // 2) projections: q = x Wq^T, k = x Wk^T, v = x Wv^T (bf16 out)
  {
    int blocks_q = (MROWS / 64) * (DM_ / 64);    // 64 * 16 = 1024
    gemm_bf16_tdm<<<blocks_q, 128, 0, stream>>>(xb, Wqb, qbw, nullptr, nullptr,
                                                MROWS, DM_, DM_);
    int blocks_kv = (MROWS / 64) * (KV_ / 64);   // 64 * 4 = 256
    gemm_bf16_tdm<<<blocks_kv, 128, 0, stream>>>(xb, Wkb, kbw, nullptr, nullptr,
                                                 MROWS, KV_, DM_);
    gemm_bf16_tdm<<<blocks_kv, 128, 0, stream>>>(xb, Wvb, vbw, nullptr, nullptr,
                                                 MROWS, KV_, DM_);
  }

  // 3) RoPE in place on q and k
  {
    int n = MROWS * (DM_ / 2);
    rope_bf16<<<(n + 255) / 256, 256, 0, stream>>>(qbw, MROWS, DM_, DM_ / 2);
    n = MROWS * (KV_ / 2);
    rope_bf16<<<(n + 255) / 256, 256, 0, stream>>>(kbw, MROWS, KV_, KV_ / 2);
  }

  // 4) transpose v for the PV operand
  {
    int n = MROWS * KV_;
    vtrans<<<(n + 255) / 256, 256, 0, stream>>>(vbw, vtw);
  }

  // 5) flash attention: 4 batches * 32 heads * 64 query tiles = 8192 waves
  attn_fwd<<<8192 / 4, 128, 0, stream>>>(qbw, kbw, vtw, atw);

  // 6) output projection: out = attn Wo^T + bo (f32 out)
  {
    int blocks = (MROWS / 64) * (DM_ / 64);
    gemm_bf16_tdm<<<blocks, 128, 0, stream>>>(atw, Wob, nullptr, (float*)d_out,
                                              bo, MROWS, DM_, DM_);
  }
}
